// GridSpatialEncoder_5540507812261
// MI455X (gfx1250) — matrix operand order repaired and checked by reading, not compiled, not run on hardware
//
#include <hip/hip_runtime.h>
#include <hip/hip_bf16.h>
#include <math.h>

// ---------------- problem constants (match reference) ----------------
#define BB     8
#define NN     2048
#define INDIM  256
#define DD     512
#define HH     4
#define HDH    128          // DD / HH
#define GXC    8
#define GYC    8
#define GC     64           // GXC*GYC
#define KC     9            // 3x3 neighborhood
#define BN     (BB * NN)        // 16384
#define BG     (BB * GC)        // 512
#define BGK    (BB * GC * KC)   // 4608
#define CWC    32.0f            // 256/8
#define CHC    32.0f
#define EPSLN  1e-5f
#define RSQRT_HD 0.08838834764831845f   // 1/sqrt(128)

typedef __attribute__((ext_vector_type(16))) __bf16 v16bf;
typedef __attribute__((ext_vector_type(8)))  float  v8f;

// =====================================================================
// GEMM  C[M,N] = A[M,K] @ W[K,N] + bias   (bf16 WMMA, f32 accumulate)
//
// Block = 256 threads = 8 waves. Block tile: 128 rows x 64 cols.
//   wave w -> rows [row0 + w*16, +16), cols [col0, col0+64) via 4 accums.
// Per K-step (32):
//   - all 256 threads stage the 32x64 W tile into LDS, fp32->bf16,
//     pre-swizzled into the exact wave32 B-fragment layout
//     (lane = col, elem i<8 -> K = hi*8 + i, i>=8 -> K = 16 + hi*8 + (i-8)), double-buffered,
//     ONE barrier per step.
//   - each wave loads its A fragment straight from global (one full
//     128B line per row per step) and issues 4 WMMAs reusing it.
// Requires M%128==0, K%32==0, N%64==0 (true for all call sites).
// =====================================================================
#define BT_HALVES (4 * 32 * 16)   // 2048 bf16 per buffer (4 KB)

__global__ void gemm_bias_wmma(const float* __restrict__ A,
                               const float* __restrict__ W,
                               const float* __restrict__ bias,
                               float* __restrict__ C,
                               int M, int K, int N) {
  __shared__ __bf16 Bs[2][BT_HALVES];

  const int wave = threadIdx.x >> 5;              // 0..7
  const int lane = threadIdx.x & 31;
  const int col0 = blockIdx.x * 64;
  const int row0 = blockIdx.y * 128 + wave * 16;  // this wave's 16 rows

  const int l  = lane & 15;                       // row (A) / col (B,C)
  const int hi = lane >> 4;                       // lane-half selector

  // staging role for this thread (fixed across K loop)
  const int sj  = threadIdx.x;
  const int st  = sj >> 6;                        // 0..3  N-subtile
  const int sr  = sj & 63;
  const int sL  = sr >> 1;                        // 0..31 target lane slot
  const int sih = (sr & 1) * 8;                   // half-index base 0 or 8
  const int scol = col0 + st * 16 + (sL & 15);    // global column
  const int skro = (sL >> 4) * 8 + sih * 2;       // K offset within tile
  __bf16* const sdst0 = &Bs[0][(st * 32 + sL) * 16 + sih];
  __bf16* const sdst1 = &Bs[1][(st * 32 + sL) * 16 + sih];

  v8f acc0 = {}, acc1 = {}, acc2 = {}, acc3 = {};
  const float* __restrict__ Arow = A + (size_t)(row0 + l) * (size_t)K;

  const int nk = K >> 5;
  for (int it = 0; it < nk; ++it) {
    const int k0 = it << 5;

    // ---- stage W[k0..k0+32) x [col0..col0+64) into LDS (bf16, swizzled)
    {
      const float* __restrict__ wp = W + (size_t)(k0 + skro) * (size_t)N + scol;
      __bf16* const dst = (it & 1) ? sdst1 : sdst0;
#pragma unroll
      for (int c = 0; c < 8; ++c) dst[c] = (__bf16)wp[(size_t)c * (size_t)N];
    }
    __syncthreads();   // double buffer => one barrier per step is sufficient

    // ---- A fragment (16x32 bf16): elems 0..7 -> K=k0+hi*8+i ; 8..15 -> +16
    const float* __restrict__ ap = Arow + k0 + hi * 8;
    __builtin_prefetch(ap + 64, 0, 0);            // next-next K block
    v16bf a;
#pragma unroll
    for (int i = 0; i < 8; ++i) {
      a[i]     = (__bf16)ap[i];
      a[i + 8] = (__bf16)ap[i + 16];
    }

    // ---- 4 WMMAs reusing the A fragment
    const __bf16* const bbuf = (it & 1) ? Bs[1] : Bs[0];
    v16bf b0 = *(const v16bf*)&bbuf[(0 * 32 + lane) * 16];
    v16bf b1 = *(const v16bf*)&bbuf[(1 * 32 + lane) * 16];
    v16bf b2 = *(const v16bf*)&bbuf[(2 * 32 + lane) * 16];
    v16bf b3 = *(const v16bf*)&bbuf[(3 * 32 + lane) * 16];
    acc0 = __builtin_amdgcn_wmma_f32_16x16x32_bf16(false, a, false, b0, (short)0, acc0, false, false);
    acc1 = __builtin_amdgcn_wmma_f32_16x16x32_bf16(false, a, false, b1, (short)0, acc1, false, false);
    acc2 = __builtin_amdgcn_wmma_f32_16x16x32_bf16(false, a, false, b2, (short)0, acc2, false, false);
    acc3 = __builtin_amdgcn_wmma_f32_16x16x32_bf16(false, a, false, b3, (short)0, acc3, false, false);
  }

  // ---- epilogue: bias + store. C/D layout: elem r -> row hi*8+r, col l.
#pragma unroll
  for (int t = 0; t < 4; ++t) {
    const int colw = col0 + t * 16 + l;
    const float bvv = bias ? bias[colw] : 0.0f;
    const v8f acc = (t == 0) ? acc0 : (t == 1) ? acc1 : (t == 2) ? acc2 : acc3;
#pragma unroll
    for (int r = 0; r < 8; ++r)
      C[(size_t)(row0 + hi * 8 + r) * (size_t)N + colw] = acc[r] + bvv;
  }
}

// ---------------- zero sums + counts (ws is poisoned, not re-zeroed) ----
__global__ void zero_kernel(float* __restrict__ sums, float* __restrict__ counts) {
  int i = blockIdx.x * blockDim.x + threadIdx.x;
  int total = BG * DD + BG;
  if (i < BG * DD) sums[i] = 0.0f;
  else if (i < total) counts[i - BG * DD] = 0.0f;
}

// ---------------- grid bucketing: segment-sum of feat per cell ----------
// thread t: point p = t>>4, chunk = t&15 handles 32 consecutive d's
__global__ void bucket_kernel(const float* __restrict__ coords,
                              const float* __restrict__ feat,
                              float* __restrict__ sums,
                              float* __restrict__ counts,
                              int* __restrict__ cell) {
  int t = blockIdx.x * blockDim.x + threadIdx.x;
  if (t >= BN * 16) return;
  int p = t >> 4, ch = t & 15;
  float x = coords[(size_t)p * 2 + 0];
  float y = coords[(size_t)p * 2 + 1];
  int gx = (int)(x / CWC); gx = gx < 0 ? 0 : (gx > GXC - 1 ? GXC - 1 : gx);
  int gy = (int)(y / CHC); gy = gy < 0 ? 0 : (gy > GYC - 1 ? GYC - 1 : gy);
  int b = p / NN;
  int c = gx * GYC + gy;
  if (ch == 0) { cell[p] = c; atomicAdd(&counts[b * GC + c], 1.0f); }
  float* __restrict__ dst = sums + ((size_t)(b * GC + c)) * DD + ch * 32;
  const float* __restrict__ src = feat + (size_t)p * DD + ch * 32;
#pragma unroll 4
  for (int j = 0; j < 32; ++j) atomicAdd(&dst[j], src[j]);
}

// ---------------- mean = sums / max(count,1)  (in place) ----------------
__global__ void mean_kernel(float* __restrict__ sums, const float* __restrict__ counts) {
  int i = blockIdx.x * blockDim.x + threadIdx.x;
  if (i >= BG * DD) return;
  float c = counts[i / DD];
  sums[i] = sums[i] / fmaxf(c, 1.0f);
}

// ---------------- positional encoding of the 9 offsets ------------------
// pe[k,:] = relu(off_k @ W_p1 + b_p1) @ W_p2 + b_p2 ; one block per k
__global__ void pe_kernel(const float* __restrict__ W_p1, const float* __restrict__ b_p1,
                          const float* __restrict__ W_p2, const float* __restrict__ b_p2,
                          float* __restrict__ pe) {
  __shared__ float hidden[256];
  int k = blockIdx.x;            // 0..8
  int j = threadIdx.x;           // 0..255
  float dx = (float)(k / 3 - 1);
  float dy = (float)(k % 3 - 1);
  hidden[j] = fmaxf(dx * W_p1[j] + dy * W_p1[256 + j] + b_p1[j], 0.0f);
  __syncthreads();
  for (int d = j; d < DD; d += 256) {
    float s = b_p2[d];
    for (int jj = 0; jj < 256; ++jj) s += hidden[jj] * W_p2[(size_t)jj * DD + d];
    pe[(size_t)k * DD + d] = s;
  }
}

// ---------------- keys[b,g,k,:] = mean[b, nbr(g,k)] + pe[k] --------------
__global__ void keys_kernel(const float* __restrict__ mean, const float* __restrict__ pe,
                            float* __restrict__ keys) {
  size_t i = (size_t)blockIdx.x * blockDim.x + threadIdx.x;
  if (i >= (size_t)BGK * DD) return;
  int d = (int)(i % DD);
  size_t r = i / DD;
  int k = (int)(r % KC);
  int bg = (int)(r / KC);
  int g = bg % GC, b = bg / GC;
  int gx = g / GYC, gy = g % GYC;
  int ngx = gx + (k / 3 - 1); ngx = ngx < 0 ? 0 : (ngx > GXC - 1 ? GXC - 1 : ngx);
  int ngy = gy + (k % 3 - 1); ngy = ngy < 0 ? 0 : (ngy > GYC - 1 ? GYC - 1 : ngy);
  int nc = ngx * GYC + ngy;
  keys[i] = mean[((size_t)(b * GC + nc)) * DD + d] + pe[(size_t)k * DD + d];
}

// ---------------- per-point attention over <=9 neighbor cells ------------
// one wave32 per (b,n,h); lane owns 4 consecutive d's of HD=128
__global__ void attn_kernel(const float* __restrict__ q,
                            const float* __restrict__ kk,
                            const float* __restrict__ vv,
                            const float* __restrict__ counts,
                            const int* __restrict__ cell,
                            float* __restrict__ agg) {
  int gw = (int)(((size_t)blockIdx.x * blockDim.x + threadIdx.x) >> 5);
  int lane = threadIdx.x & 31;
  if (gw >= BN * HH) return;
  int h = gw % HH;
  int pn = gw / HH;                 // point index in [0, B*N)
  int b = pn / NN;
  int c = cell[pn];
  int cgx = c / GYC, cgy = c % GYC;

  const float4 qv = *(const float4*)(q + (size_t)pn * DD + h * HDH + lane * 4);

  float sc[KC];
  size_t rowbase = (size_t)(b * GC + c) * KC;
#pragma unroll
  for (int k = 0; k < KC; ++k) {
    int dx = k / 3 - 1, dy = k % 3 - 1;
    int ngx = cgx + dx, ngy = cgy + dy;
    bool inb = (ngx >= 0) & (ngx < GXC) & (ngy >= 0) & (ngy < GYC);
    int cx = ngx < 0 ? 0 : (ngx > GXC - 1 ? GXC - 1 : ngx);
    int cy = ngy < 0 ? 0 : (ngy > GYC - 1 ? GYC - 1 : ngy);
    bool valid = inb && (counts[b * GC + cx * GYC + cy] > 0.0f);
    const float4 kv = *(const float4*)(kk + (rowbase + k) * DD + h * HDH + lane * 4);
    float s = qv.x * kv.x + qv.y * kv.y + qv.z * kv.z + qv.w * kv.w;
#pragma unroll
    for (int off = 16; off > 0; off >>= 1) s += __shfl_xor(s, off, 32);
    sc[k] = valid ? s * RSQRT_HD : -1e9f;
  }
  // softmax over 9 (redundant on all lanes)
  float mx = sc[0];
#pragma unroll
  for (int k = 1; k < KC; ++k) mx = fmaxf(mx, sc[k]);
  float den = 0.0f, w[KC];
#pragma unroll
  for (int k = 0; k < KC; ++k) { w[k] = __expf(sc[k] - mx); den += w[k]; }
  float inv = 1.0f / den;

  float4 out = {0.f, 0.f, 0.f, 0.f};
#pragma unroll
  for (int k = 0; k < KC; ++k) {
    const float4 vvv = *(const float4*)(vv + (rowbase + k) * DD + h * HDH + lane * 4);
    float a = w[k] * inv;
    out.x += a * vvv.x; out.y += a * vvv.y; out.z += a * vvv.z; out.w += a * vvv.w;
  }
  *(float4*)(agg + (size_t)pn * DD + h * HDH + lane * 4) = out;
}

// ---------------- residual + LayerNorm (one wave32 per point) ------------
__global__ void ln_kernel(const float* __restrict__ feat, const float* __restrict__ proj,
                          const float* __restrict__ gamma, const float* __restrict__ beta,
                          float* __restrict__ out) {
  int gw = (int)(((size_t)blockIdx.x * blockDim.x + threadIdx.x) >> 5);
  int lane = threadIdx.x & 31;
  if (gw >= BN) return;
  const float* __restrict__ f = feat + (size_t)gw * DD;
  const float* __restrict__ p = proj + (size_t)gw * DD;
  float e[16];
  float s = 0.0f, s2 = 0.0f;
#pragma unroll
  for (int i = 0; i < 16; ++i) {
    float v = f[lane + i * 32] + p[lane + i * 32];
    e[i] = v; s += v; s2 += v * v;
  }
#pragma unroll
  for (int off = 16; off > 0; off >>= 1) {
    s  += __shfl_xor(s,  off, 32);
    s2 += __shfl_xor(s2, off, 32);
  }
  float mu  = s * (1.0f / DD);
  float var = s2 * (1.0f / DD) - mu * mu;
  float rs  = rsqrtf(var + EPSLN);
  float* __restrict__ o = out + (size_t)gw * DD;
#pragma unroll
  for (int i = 0; i < 16; ++i) {
    int d = lane + i * 32;
    o[d] = (e[i] - mu) * rs * gamma[d] + beta[d];
  }
}

// =====================================================================
extern "C" void kernel_launch(void* const* d_in, const int* in_sizes, int n_in,
                              void* d_out, int out_size, void* d_ws, size_t ws_size,
                              hipStream_t stream) {
  const float* features = (const float*)d_in[0];
  const float* coords   = (const float*)d_in[1];
  const float* W_feat   = (const float*)d_in[2];
  const float* b_feat   = (const float*)d_in[3];
  const float* W_p1     = (const float*)d_in[4];
  const float* b_p1     = (const float*)d_in[5];
  const float* W_p2     = (const float*)d_in[6];
  const float* b_p2     = (const float*)d_in[7];
  const float* Wq       = (const float*)d_in[8];
  const float* bq       = (const float*)d_in[9];
  const float* Wk       = (const float*)d_in[10];
  const float* bk       = (const float*)d_in[11];
  const float* Wv       = (const float*)d_in[12];
  const float* bv       = (const float*)d_in[13];
  const float* Wo       = (const float*)d_in[14];
  const float* bo       = (const float*)d_in[15];
  const float* gamma    = (const float*)d_in[16];
  const float* beta     = (const float*)d_in[17];
  float* outp = (float*)d_out;

  // ---- workspace layout (floats) ----
  float* ws = (float*)d_ws;
  size_t off = 0;
  float* feat  = ws + off; off += (size_t)BN * DD;    // kept for residual
  float* qbuf  = ws + off; off += (size_t)BN * DD;    // q, later reused as proj
  float* agg   = ws + off; off += (size_t)BN * DD;    // attention output
  float* sums  = ws + off; off += (size_t)BG * DD;    // becomes mean in place
  float* cnts  = ws + off; off += (size_t)BG;
  float* pe    = ws + off; off += (size_t)KC * DD;
  float* keys  = ws + off; off += (size_t)BGK * DD;
  float* kk    = ws + off; off += (size_t)BGK * DD;
  float* vv    = ws + off; off += (size_t)BGK * DD;
  int*   cell  = (int*)(ws + off); off += (size_t)BN;
  if (ws_size < off * sizeof(float)) return;          // insufficient scratch

  const dim3 blk(256);
  auto cdiv = [](size_t a, size_t b) { return (unsigned)((a + b - 1) / b); };

  // 0) zero segment sums + counts (ws not re-zeroed between replays)
  zero_kernel<<<cdiv(BG * DD + BG, 256), blk, 0, stream>>>(sums, cnts);

  // 1) feat = features @ W_feat + b_feat   [16384,256]x[256,512]
  gemm_bias_wmma<<<dim3(DD / 64, BN / 128), blk, 0, stream>>>(
      features, W_feat, b_feat, feat, BN, INDIM, DD);

  // 2) bucket points into cells, segment-sum features
  bucket_kernel<<<cdiv((size_t)BN * 16, 256), blk, 0, stream>>>(
      coords, feat, sums, cnts, cell);

  // 3) mean per cell
  mean_kernel<<<cdiv((size_t)BG * DD, 256), blk, 0, stream>>>(sums, cnts);

  // 4) positional encoding of 9 offsets
  pe_kernel<<<dim3(KC), blk, 0, stream>>>(W_p1, b_p1, W_p2, b_p2, pe);

  // 5) keys[b,g,k] = mean[b,nbr(g,k)] + pe[k]   (only 4608 distinct rows!)
  keys_kernel<<<cdiv((size_t)BGK * DD, 256), blk, 0, stream>>>(sums, pe, keys);

  // 6) k/v projections on the deduped key rows  [4608,512]x[512,512]
  gemm_bias_wmma<<<dim3(DD / 64, BGK / 128), blk, 0, stream>>>(keys, Wk, bk, kk, BGK, DD, DD);
  gemm_bias_wmma<<<dim3(DD / 64, BGK / 128), blk, 0, stream>>>(keys, Wv, bv, vv, BGK, DD, DD);

  // 7) q = feat @ Wq + bq   [16384,512]x[512,512]
  gemm_bias_wmma<<<dim3(DD / 64, BN / 128), blk, 0, stream>>>(feat, Wq, bq, qbuf, BN, DD, DD);

  // 8) per-point attention over <=9 neighbor cells (wave per (b,n,h))
  attn_kernel<<<cdiv((size_t)BN * HH * 32, 256), blk, 0, stream>>>(
      qbuf, kk, vv, cnts, cell, agg);

  // 9) output projection: proj = agg @ Wo + bo  (reuse qbuf)
  gemm_bias_wmma<<<dim3(DD / 64, BN / 128), blk, 0, stream>>>(agg, Wo, bo, qbuf, BN, DD, DD);

  // 10) residual + LayerNorm -> d_out
  ln_kernel<<<cdiv((size_t)BN * 32, 256), blk, 0, stream>>>(feat, qbuf, gamma, beta, outp);
}
